// ConvQNN_10204842295553
// MI455X (gfx1250) — matrix operand order repaired and checked by reading, not compiled, not run on hardware
//
#include <hip/hip_runtime.h>
#include <hip/hip_bf16.h>
#include <math.h>

// ---------------------------------------------------------------------------
// Pipeline (all on-device, workspace layout):
//   [0]                 A_f16   : [8192][992] f16, conv+relu+pool output, zero-pad k>=980
//   [OFF_B]             B_f16   : fc1_w swizzled into k-tiles: Bs[(kt*256+n)*32 + ki]
//   [OFF_C]             C_f32   : [8192][256] FC1 GEMM output (pre-bias, pre-leaky)
// Kernels:
//   1) conv_pool_kernel : conv(5x5,pad2)+bias+relu+maxpool2x2 -> A_f16
//   2) wcvt_kernel      : fc1_w f32 -> swizzled f16 B tiles (zero pad)
//   3) fc1_wmma_kernel  : TDM A-tile -> LDS, WMMA f16 GEMM (M=8192,N=256,K=992)
//   4) qnn_kernel       : bias+leaky+normalize+8-qubit sim+<Z>+FC2 -> out
// ---------------------------------------------------------------------------

typedef __attribute__((ext_vector_type(16))) _Float16 v16h;
typedef __attribute__((ext_vector_type(8)))  _Float16 v8h;
typedef __attribute__((ext_vector_type(8)))  float    v8f;

#define BATCH   8192
#define KPAD    992           // 980 padded to 31*32
#define KTILES  31
#define NOUT    256

#if defined(__HIP_DEVICE_COMPILE__) && __has_builtin(__builtin_amdgcn_tensor_load_to_lds)
#define HAVE_TDM 1
typedef unsigned int u32x4 __attribute__((ext_vector_type(4)));
typedef int          i32x4 __attribute__((ext_vector_type(4)));
typedef int          i32x8 __attribute__((ext_vector_type(8)));
#else
#define HAVE_TDM 0
#endif

// ------------------------------ kernel 1 -----------------------------------
// one thread per (sample, padded column). col < 980: fused conv+relu+pool.
__global__ __launch_bounds__(256) void conv_pool_kernel(
    const float* __restrict__ x,       // [8192,1,28,28]
    const float* __restrict__ w,       // [5,1,5,5]
    const float* __restrict__ bias,    // [5]
    _Float16* __restrict__ A)          // [8192,992]
{
    int t = blockIdx.x * 256 + threadIdx.x;     // 8192*992 threads exactly
    int col = t % KPAD;
    int s   = t / KPAD;
    if (col >= 980) { A[t] = (_Float16)0.f; return; }

    int c  = col / 196;
    int ph = (col % 196) / 14;
    int pw = col % 14;

    const float* xs = x + s * 784;
    const float* wc = w + c * 25;
    float bc = bias[c];

    float best = -1e30f;
    #pragma unroll
    for (int dy = 0; dy < 2; ++dy) {
        #pragma unroll
        for (int dx = 0; dx < 2; ++dx) {
            int oy = 2 * ph + dy;
            int ox = 2 * pw + dx;
            float acc = bc;
            #pragma unroll
            for (int ky = 0; ky < 5; ++ky) {
                int iy = oy + ky - 2;
                if (iy < 0 || iy > 27) continue;
                #pragma unroll
                for (int kx = 0; kx < 5; ++kx) {
                    int ix = ox + kx - 2;
                    if (ix < 0 || ix > 27) continue;
                    acc += xs[iy * 28 + ix] * wc[ky * 5 + kx];
                }
            }
            acc = fmaxf(acc, 0.f);            // relu (bias pre-applied)
            best = fmaxf(best, acc);          // maxpool
        }
    }
    A[s * KPAD + col] = (_Float16)best;
}

// ------------------------------ kernel 2 -----------------------------------
// fc1_w [256,980] f32 -> swizzled f16: Bs[(kt*256 + n)*32 + ki] = W[n][kt*32+ki]
__global__ __launch_bounds__(256) void wcvt_kernel(
    const float* __restrict__ w1,      // [256,980]
    _Float16* __restrict__ Bs)         // [31*256*32]
{
    int t = blockIdx.x * 256 + threadIdx.x;     // 992*256 threads exactly
    int k = t / NOUT;
    int n = t % NOUT;
    int kt = k / 32, ki = k % 32;
    float v = (k < 980) ? w1[n * 980 + k] : 0.f;
    Bs[(kt * NOUT + n) * 32 + ki] = (_Float16)v;
}

// ------------------------------ kernel 3 -----------------------------------
// WMMA GEMM: C[M=8192][N=256] = A[M][K=992] * B[K][N].
// Block = 4 waves; wave w owns n-tile (blockIdx.y*64 + w*16), blockIdx.x owns m-tile.
// The block's 16x992 A tile (31,744 B) is DMA'd into LDS once by the Tensor
// Data Mover (2D descriptor, data_size=2B), then all four waves feed their A
// fragments from LDS (ds_load_b128) -- removes the 4x redundant global A reads.
//
// NOTE: the TDM builtin's LDS writes are invisible to alias analysis (the
// descriptor carries lds_addr as an integer). Without a visible store, LLVM
// folds every load of As to undef (observed in round 2). We prime As with a
// single store through an opaque (inline-asm-produced) index so no LDS load
// can be proven dead, and order it before the TDM with s_wait_dscnt.
__global__ __launch_bounds__(128) void fc1_wmma_kernel(
    const _Float16* __restrict__ A,    // [8192][992]
    const _Float16* __restrict__ Bs,   // swizzled tiles
    float* __restrict__ C)             // [8192][256]
{
    __shared__ __align__(16) _Float16 As[16 * KPAD];   // 31,744 bytes, LDS offset 0

    const int lane = threadIdx.x & 31;
    const int wv   = threadIdx.x >> 5;
    const int m0   = blockIdx.x * 16;
    const int n0   = blockIdx.y * 64 + wv * 16;

#if HAVE_TDM
    if (wv == 0) {
        if (lane == 0) {
            // Opaque index (value 0 at runtime): compiler must keep all As loads.
            unsigned pidx;
            asm volatile("v_mov_b32 %0, 0" : "=v"(pidx));
            As[pidx] = (_Float16)0.f;
        }
        // Priming store must land before the TDM overwrites it (TDM is
        // unordered w.r.t. DS instructions).
        asm volatile("s_wait_dscnt 0x0" ::: "memory");

        // ---- Tensor DMA descriptor (ISA 8.3/8.4), 2D: 16 rows x 992 halves ----
        unsigned long long ga =
            (unsigned long long)(const void*)(A + (size_t)m0 * KPAD);
        u32x4 g0;
        g0[0] = 1u;                                        // count=1 (valid user D#)
        g0[1] = 0u;                                        // lds_addr = 0 (As)
        g0[2] = (unsigned)(ga & 0xffffffffu);              // global_addr[31:0]
        g0[3] = (unsigned)((ga >> 32) & 0x01ffffffu)       // global_addr[56:32]
              | (2u << 30);                                // type = 2 ("image")
        i32x8 g1;
        g1[0] = (int)(1u << 16);        // workgroup_mask=0, data_size=1 (2 bytes)
        g1[1] = (int)(992u << 16);      // tensor_dim0 = 992 (lo16 at [31:16])
        g1[2] = (int)(16u << 16);       // dim0 hi=0 | tensor_dim1 = 16
        g1[3] = (int)(992u << 16);      // dim1 hi=0 | tile_dim0 = 992
        g1[4] = 16;                     // tile_dim1 = 16, tile_dim2 = 0
        g1[5] = 992;                    // tensor_dim0_stride[31:0] = 992
        g1[6] = 0;                      // stride hi | tensor_dim1_stride lo
        g1[7] = 0;
        i32x4 z4 = {0, 0, 0, 0};        // groups 2/3 unused (2D tensor)
#if __clang_major__ >= 23
        i32x8 z8 = {0, 0, 0, 0, 0, 0, 0, 0};
        __builtin_amdgcn_tensor_load_to_lds(g0, g1, z4, z4, z8, 0);
#else
        __builtin_amdgcn_tensor_load_to_lds(g0, g1, z4, z4, 0);
#endif
#if __has_builtin(__builtin_amdgcn_s_wait_tensorcnt)
        __builtin_amdgcn_s_wait_tensorcnt(0);
#else
        asm volatile("s_wait_tensorcnt 0x0" ::: "memory");
#endif
        asm volatile("" ::: "memory");
    }
    __syncthreads();
#else
    // Fallback: cooperative global->LDS copy in 16-byte chunks.
    {
        const v8h* src = (const v8h*)(A + (size_t)m0 * KPAD);
        v8h* dst = (v8h*)As;
        for (int i = threadIdx.x; i < (16 * KPAD) / 8; i += 128) dst[i] = src[i];
    }
    __syncthreads();
#endif

    const int hiH   = lane >> 4;             // 0 for lanes 0-15, 1 for 16-31
    const int abase = hiH * 8;               // A: K offset within 32-wide tile
    const int col   = n0 + (lane & 15);
    const int bbase = hiH * 8;               // B: K offset within 32-wide tile

    const _Float16* arow = As + (lane & 15) * KPAD;

    v8f acc = {};
    #pragma unroll 4
    for (int kt = 0; kt < KTILES; ++kt) {
        const int k0 = kt * 32;
        // A fragment (from LDS): halves 0..7 = K(base..base+7), 8..15 = K(16+base..)
        v8h alo = *(const v8h*)(arow + k0 + abase);
        v8h ahi = *(const v8h*)(arow + k0 + abase + 16);
        // B fragment (global/L2): halves 0..7 = K(base..base+7), 8..15 = K(16+base..)
        const _Float16* bcol = Bs + (kt * NOUT + col) * 32 + bbase;
        v8h blo = *(const v8h*)(bcol);
        v8h bhi = *(const v8h*)(bcol + 16);

        v16h a, b;
        #pragma unroll
        for (int i = 0; i < 8; ++i) {
            a[i] = alo[i]; a[i + 8] = ahi[i];
            b[i] = blo[i]; b[i + 8] = bhi[i];
        }
        acc = __builtin_amdgcn_wmma_f32_16x16x32_f16(
                  false, a, false, b, (short)0, acc, false, false);
    }

    // C/D layout: VGPR v holds M = v + 8*(lane>=16), N = lane&15
    const int nO = n0 + (lane & 15);
    const int mB = m0 + hiH * 8;
    #pragma unroll
    for (int v = 0; v < 8; ++v)
        C[(mB + v) * NOUT + nO] = acc[v];
}

// ------------------------------ kernel 4 -----------------------------------
// One wave32 per sample. Amplitude index i in [0,256):
//   lane = i>>3  (bits = wires 0..4, wire0 = lane bit4 .. wire4 = lane bit0)
//   r    = i&7   (bits = wires 5..7, wire5 = r bit2 .. wire7 = r bit0)
__global__ __launch_bounds__(256) void qnn_kernel(
    const float* __restrict__ C,       // [8192,256] FC1 pre-activation
    const float* __restrict__ b1,      // [256]
    const float* __restrict__ qp,      // [4,8,3]
    const float* __restrict__ w2,      // [10,8]
    const float* __restrict__ b2,      // [10]
    float* __restrict__ out)           // [8192,10]
{
    const int lane = threadIdx.x & 31;
    const int wv   = threadIdx.x >> 5;
    const int s    = blockIdx.x * 8 + wv;

    // ---- load, bias, leaky-relu ----
    float h[8];
    const float* hrow = C  + s * 256 + lane * 8;
    const float* brow = b1 + lane * 8;
    #pragma unroll
    for (int r = 0; r < 8; ++r) {
        float t = hrow[r] + brow[r];
        h[r] = t > 0.f ? t : 0.01f * t;
    }
    // ---- two-step normalization (matches reference) ----
    float S = 0.f;
    #pragma unroll
    for (int r = 0; r < 8; ++r) S += h[r] * h[r];
    #pragma unroll
    for (int m = 16; m >= 1; m >>= 1) S += __shfl_xor(S, m, 32);
    float inv1 = 1.f / (sqrtf(S) + 1e-8f);
    #pragma unroll
    for (int r = 0; r < 8; ++r) h[r] *= inv1;
    float S2 = 0.f;
    #pragma unroll
    for (int r = 0; r < 8; ++r) S2 += h[r] * h[r];
    #pragma unroll
    for (int m = 16; m >= 1; m >>= 1) S2 += __shfl_xor(S2, m, 32);
    float inv2 = 1.f / sqrtf(S2);

    float re[8], im[8];
    #pragma unroll
    for (int r = 0; r < 8; ++r) { re[r] = h[r] * inv2; im[r] = 0.f; }

    // ---- 4 layers: 8x U3 + ring CNOTs ----
    for (int layer = 0; layer < 4; ++layer) {
        #pragma unroll
        for (int j = 0; j < 8; ++j) {
            const float th = qp[(layer * 8 + j) * 3 + 0];
            const float ph = qp[(layer * 8 + j) * 3 + 1];
            const float lm = qp[(layer * 8 + j) * 3 + 2];
            float cth = cosf(0.5f * th), sth = sinf(0.5f * th);
            float cl = cosf(lm), sl = sinf(lm);
            float cp = cosf(ph), sp = sinf(ph);
            float cpl = cp * cl - sp * sl, spl = sp * cl + cp * sl;
            // U3 = [[c, -e^{il} s],[e^{ip} s, e^{i(p+l)} c]]
            float u00r = cth,       u00i = 0.f;
            float u01r = -cl * sth, u01i = -sl * sth;
            float u10r = cp * sth,  u10i = sp * sth;
            float u11r = cpl * cth, u11i = spl * cth;

            if (j >= 5) {                      // register wire
                const int rb = 1 << (7 - j);
                #pragma unroll
                for (int r = 0; r < 8; ++r) {
                    if (r & rb) continue;
                    const int r1 = r | rb;
                    float ar = re[r],  ai = im[r];
                    float br = re[r1], bi = im[r1];
                    re[r]  = u00r*ar - u00i*ai + u01r*br - u01i*bi;
                    im[r]  = u00r*ai + u00i*ar + u01r*bi + u01i*br;
                    re[r1] = u10r*ar - u10i*ai + u11r*br - u11i*bi;
                    im[r1] = u10r*ai + u10i*ar + u11r*bi + u11i*br;
                }
            } else {                           // lane wire -> shuffle butterfly
                const int lb = 1 << (4 - j);
                const bool hi = (lane & lb) != 0;
                float c0r = hi ? u10r : u00r, c0i = hi ? u10i : u00i;
                float c1r = hi ? u11r : u01r, c1i = hi ? u11i : u01i;
                #pragma unroll
                for (int r = 0; r < 8; ++r) {
                    float pr = __shfl_xor(re[r], lb, 32);
                    float pi = __shfl_xor(im[r], lb, 32);
                    float ar = hi ? pr : re[r], ai = hi ? pi : im[r];
                    float br = hi ? re[r] : pr, bi = hi ? im[r] : pi;
                    re[r] = c0r*ar - c0i*ai + c1r*br - c1i*bi;
                    im[r] = c0r*ai + c0i*ar + c1r*bi + c1i*br;
                }
            }
        }
        // CNOT(c, c+1) for c=0..3 : both wires in lane bits
        #pragma unroll
        for (int c = 0; c < 4; ++c) {
            const int cb = 1 << (4 - c);
            const int tb = 1 << (3 - c);
            const bool ctl = (lane & cb) != 0;
            #pragma unroll
            for (int r = 0; r < 8; ++r) {
                float pr = __shfl_xor(re[r], tb, 32);
                float pi = __shfl_xor(im[r], tb, 32);
                re[r] = ctl ? pr : re[r];
                im[r] = ctl ? pi : im[r];
            }
        }
        // CNOT(4,5): ctrl = lane bit0, tgt = reg bit2 -> conditional reg swap
        {
            const bool ctl = (lane & 1) != 0;
            #pragma unroll
            for (int r = 0; r < 4; ++r) {
                float tr = re[r], ti = im[r];
                re[r]   = ctl ? re[r+4] : re[r];
                im[r]   = ctl ? im[r+4] : im[r];
                re[r+4] = ctl ? tr : re[r+4];
                im[r+4] = ctl ? ti : im[r+4];
            }
        }
        // CNOT(5,6): regs with r&4 swap r <-> r^2  : (4,6),(5,7)
        { float t;
          t=re[4]; re[4]=re[6]; re[6]=t;  t=im[4]; im[4]=im[6]; im[6]=t;
          t=re[5]; re[5]=re[7]; re[7]=t;  t=im[5]; im[5]=im[7]; im[7]=t; }
        // CNOT(6,7): regs with r&2 swap r <-> r^1  : (2,3),(6,7)
        { float t;
          t=re[2]; re[2]=re[3]; re[3]=t;  t=im[2]; im[2]=im[3]; im[3]=t;
          t=re[6]; re[6]=re[7]; re[7]=t;  t=im[6]; im[6]=im[7]; im[7]=t; }
        // CNOT(7,0): ctrl = reg bit0, tgt = lane bit4 -> shuffle odd regs
        #pragma unroll
        for (int r = 1; r < 8; r += 2) {
            re[r] = __shfl_xor(re[r], 16, 32);
            im[r] = __shfl_xor(im[r], 16, 32);
        }
    }

    // ---- <Z_j> ----
    float p[8], Sp = 0.f;
    #pragma unroll
    for (int r = 0; r < 8; ++r) { p[r] = re[r]*re[r] + im[r]*im[r]; Sp += p[r]; }
    float z[8];
    #pragma unroll
    for (int j = 0; j < 5; ++j) {
        const int lb = 1 << (4 - j);
        z[j] = (lane & lb) ? -Sp : Sp;
    }
    z[5] = (p[0]+p[1]+p[2]+p[3]) - (p[4]+p[5]+p[6]+p[7]);
    z[6] = (p[0]+p[1]+p[4]+p[5]) - (p[2]+p[3]+p[6]+p[7]);
    z[7] = (p[0]+p[2]+p[4]+p[6]) - (p[1]+p[3]+p[5]+p[7]);
    #pragma unroll
    for (int j = 0; j < 8; ++j) {
        #pragma unroll
        for (int m = 16; m >= 1; m >>= 1) z[j] += __shfl_xor(z[j], m, 32);
    }

    // ---- FC2 ----
    if (lane < 10) {
        float o = b2[lane];
        #pragma unroll
        for (int j = 0; j < 8; ++j) o += z[j] * w2[lane * 8 + j];
        out[s * 10 + lane] = o;
    }
}

// ------------------------------ launcher -----------------------------------
extern "C" void kernel_launch(void* const* d_in, const int* in_sizes, int n_in,
                              void* d_out, int out_size, void* d_ws, size_t ws_size,
                              hipStream_t stream) {
    const float* x       = (const float*)d_in[0];
    const float* conv1_w = (const float*)d_in[1];
    const float* conv1_b = (const float*)d_in[2];
    const float* fc1_w   = (const float*)d_in[3];
    const float* fc1_b   = (const float*)d_in[4];
    const float* qnn_p   = (const float*)d_in[5];
    const float* fc2_w   = (const float*)d_in[6];
    const float* fc2_b   = (const float*)d_in[7];
    float* out = (float*)d_out;

    char* ws = (char*)d_ws;
    const size_t OFF_B = (size_t)BATCH * KPAD * sizeof(_Float16);        // 16,252,928
    const size_t OFF_C = OFF_B + (size_t)KPAD * NOUT * sizeof(_Float16); // +507,904
    _Float16* A  = (_Float16*)(ws);
    _Float16* Bs = (_Float16*)(ws + OFF_B);
    float*    Cm = (float*)   (ws + OFF_C);

    // 1) conv + relu + maxpool -> f16 padded activations
    conv_pool_kernel<<<(BATCH * KPAD) / 256, 256, 0, stream>>>(x, conv1_w, conv1_b, A);

    // 2) weight convert + swizzle
    wcvt_kernel<<<(KPAD * NOUT) / 256, 256, 0, stream>>>(fc1_w, Bs);

    // 3) TDM-staged WMMA GEMM
    dim3 g(BATCH / 16, NOUT / 64);
    fc1_wmma_kernel<<<g, 128, 0, stream>>>(A, Bs, Cm);

    // 4) bias + leaky + normalize + quantum sim + <Z> + FC2
    qnn_kernel<<<BATCH / 8, 256, 0, stream>>>(Cm, fc1_b, qnn_p, fc2_w, fc2_b, out);
}
